// VectorAttention_42588895707399
// MI455X (gfx1250) — matrix operand order repaired and checked by reading, not compiled, not run on hardware
//
#include <hip/hip_runtime.h>
#include <hip/hip_bf16.h>
#include <math.h>
#include <stdint.h>

typedef __attribute__((ext_vector_type(16))) __bf16 v16bf;
typedef __attribute__((ext_vector_type(8)))  __bf16 v8bf;
typedef __attribute__((ext_vector_type(8)))  float  v8f;

#define NPTS 1024
#define CCH  128
#define HEADS 4
#define DHEAD 32

// Build a 16-element A fragment (16-bit A-matrix 16x32 layout):
// lane half h: K chunks [8h, 8h+8) and [16+8h, 16+8h+8)
__device__ __forceinline__ v16bf load_afrag(const __bf16* row, int half) {
    v8bf a0 = *(const v8bf*)(row + 8 * half);
    v8bf a1 = *(const v8bf*)(row + 16 + 8 * half);
    return __builtin_shufflevector(a0, a1, 0,1,2,3,4,5,6,7,8,9,10,11,12,13,14,15);
}
// B fragment (32x16): lane half h: K chunks [8h, 8h+8) and [16+8h, 16+8h+8) of column (lane&15)
__device__ __forceinline__ v16bf load_bfrag(const __bf16* col, int half) {
    v8bf b0 = *(const v8bf*)(col + 8 * half);
    v8bf b1 = *(const v8bf*)(col + 16 + 8 * half);
    return __builtin_shufflevector(b0, b1, 0,1,2,3,4,5,6,7,8,9,10,11,12,13,14,15);
}
__device__ __forceinline__ v8f wmma_bf16(v16bf a, v16bf b, v8f c) {
    return __builtin_amdgcn_wmma_f32_16x16x32_bf16(false, a, false, b, (short)0, c, false, false);
}

// CDNA5 async memory->LDS copy (ASYNCcnt-tracked). ldsoff = wave LDS byte offset,
// goff = byte offset from 64-bit base in SGPRs (GVS addressing).
__device__ __forceinline__ void async_ld_b128(uint32_t ldsoff, uint32_t goff, const void* base) {
    asm volatile("global_load_async_to_lds_b128 %0, %1, %2 offset:0"
                 :: "v"(ldsoff), "v"(goff), "s"(base) : "memory");
}
__device__ __forceinline__ void wait_asynccnt0() {
    asm volatile("s_wait_asynccnt 0" ::: "memory");
}

// ---------------- prep kernels ----------------

// Convert Wq,Wk,Wv,Wo (each 128x128 fp32) to contiguous bf16 [4][16384]
__global__ __launch_bounds__(256) void k_cvt_w(const float* wq, const float* wk,
                                               const float* wv, const float* wo,
                                               __bf16* out) {
    int i = blockIdx.x * 256 + threadIdx.x;             // 0..65535
    if (i >= 4 * CCH * CCH) return;
    const float* src = (i < 16384) ? wq : (i < 32768) ? wk : (i < 49152) ? wv : wo;
    out[i] = (__bf16)src[i & 16383];
}

// xt[n][c] = bf16(x[c][n]);  P[c][n] = Wp1[c,:] . xyz[n,:]
__global__ __launch_bounds__(256) void k_xtP(const float* x, const float* xyz,
                                             const float* Wp1, __bf16* xt, float* P) {
    int i = blockIdx.x * 256 + threadIdx.x;             // 0..131071
    if (i >= CCH * NPTS) return;
    int c = i >> 10, n = i & (NPTS - 1);
    xt[n * CCH + c] = (__bf16)x[i];
    P[i] = Wp1[c * 3 + 0] * xyz[n * 3 + 0]
         + Wp1[c * 3 + 1] * xyz[n * 3 + 1]
         + Wp1[c * 3 + 2] * xyz[n * 3 + 2];
}

// wbar[h][c] = mean_d Wp2[h*32+d][c];  bbar[h] = mean_d bp2[h*32+d]
__global__ __launch_bounds__(256) void k_wbar(const float* Wp2, const float* bp2,
                                              float* wbar, float* bbar) {
    int idx = blockIdx.x * 256 + threadIdx.x;
    if (idx < HEADS * CCH) {
        int h = idx >> 7, c = idx & 127;
        float s = 0.f;
        for (int d = 0; d < DHEAD; ++d) s += Wp2[(h * DHEAD + d) * CCH + c];
        wbar[idx] = s * (1.f / 32.f);
    }
    if (idx < HEADS) {
        float s = 0.f;
        for (int d = 0; d < DHEAD; ++d) s += bp2[idx * DHEAD + d];
        bbar[idx] = s * (1.f / 32.f);
    }
}

// ---------------- QKV projection GEMM (WMMA bf16) ----------------
// Out[o,n] = sum_c W[o,c] x[c,n].  A = W row-major, B = xt[n][c] (c-contiguous).
// z=0: Q -> qh[h][n][d], z=1: K -> kh[h][n][d], z=2: V -> vb[c][n]
__global__ __launch_bounds__(32) void k_gemm_qkv(const __bf16* wq, const __bf16* wk,
                                                 const __bf16* wv, const __bf16* xt,
                                                 __bf16* qh, __bf16* kh, __bf16* vb) {
    int which = blockIdx.z;
    const __bf16* W = (which == 0) ? wq : (which == 1) ? wk : wv;
    int ot = blockIdx.x, nt = blockIdx.y;
    int lane = threadIdx.x, half = lane >> 4, lo = lane & 15;
    int arow = ot * 16 + lo;
    int brow = nt * 16 + lo;
    v8f acc = {};
    #pragma unroll
    for (int k0 = 0; k0 < CCH; k0 += 32) {
        v16bf a = load_afrag(W + arow * CCH + k0, half);
        v16bf b = load_bfrag(xt + brow * CCH + k0, half);
        acc = wmma_bf16(a, b, acc);
    }
    int n = nt * 16 + lo;
    int obase = ot * 16 + 8 * half;
    if (which == 2) {                                    // V as [c][n] (B operand for attn.V)
        #pragma unroll
        for (int v = 0; v < 8; ++v)
            vb[(obase + v) * NPTS + n] = (__bf16)acc[v];
    } else {                                             // Q/K as [h][pos][d]
        __bf16* dst = (which == 0) ? qh : kh;
        #pragma unroll
        for (int v = 0; v < 8; ++v) {
            int o = obase + v;
            dst[((o >> 5) * NPTS + n) * DHEAD + (o & 31)] = (__bf16)acc[v];
        }
    }
}

// ---------------- fused scores: QK^T (WMMA) + gelu positional bias ----------------
// scores[h][n][m] = (Q^T K)/sqrt(D) + sum_c wbar[h][c]*gelu(P[c,n]-P[c,m]+bp1[c]) + bbar[h]
__global__ __launch_bounds__(32) void k_scores(const __bf16* qh, const __bf16* kh,
                                               const float* P, const float* bp1,
                                               const float* wbar, const float* bbar,
                                               float* scores) {
    __shared__ __align__(16) float sPn[CCH][16];   // raw P[c][n-tile]
    __shared__ __align__(16) float sPm[CCH][16];   // raw P[c][m-tile]
    __shared__ float sW[HEADS][CCH];
    __shared__ float sb1[CCH];
    int nt = blockIdx.x, mt = blockIdx.y;
    int lane = threadIdx.x, half = lane >> 4, lo = lane & 15;
    int n0 = nt * 16, m0 = mt * 16;

    // Stage P tiles via CDNA5 async memory->LDS (16B per lane per op; 32 ops total)
    uint32_t ldsPn = (uint32_t)(uintptr_t)&sPn[0][0];
    uint32_t ldsPm = (uint32_t)(uintptr_t)&sPm[0][0];
    #pragma unroll
    for (int it = 0; it < 16; ++it) {
        int idx = it * 128 + lane * 4;                 // float index within 128x16 tile
        int c = idx >> 4, col = idx & 15;
        uint32_t goffn = (uint32_t)((c * NPTS + n0 + col) * 4);
        uint32_t goffm = (uint32_t)((c * NPTS + m0 + col) * 4);
        async_ld_b128(ldsPn + (uint32_t)idx * 4, goffn, P);
        async_ld_b128(ldsPm + (uint32_t)idx * 4, goffm, P);
    }
    for (int idx = lane; idx < HEADS * CCH; idx += 32)
        sW[idx >> 7][idx & 127] = wbar[idx];
    for (int idx = lane; idx < CCH; idx += 32)
        sb1[idx] = bp1[idx];

    // QK^T while the async copy is in flight: one 16x16x32 WMMA per head (D=32)
    v8f acc[HEADS];
    #pragma unroll
    for (int h = 0; h < HEADS; ++h) {
        const __bf16* qrow = qh + (size_t)(h * NPTS + n0 + lo) * DHEAD;
        const __bf16* krow = kh + (size_t)(h * NPTS + m0 + lo) * DHEAD;
        v16bf a = load_afrag(qrow, half);
        v16bf b = load_bfrag(krow, half);
        v8f z = {};
        acc[h] = wmma_bf16(a, b, z);
    }

    wait_asynccnt0();
    __syncthreads();

    // positional bias: exact gelu, shared across all 4 heads
    float pos0[8] = {}, pos1[8] = {}, pos2[8] = {}, pos3[8] = {};
    for (int c = 0; c < CCH; ++c) {
        float pmb = sPm[c][lo] - sb1[c];               // fold +bp1[c] into the subtrahend
        float w0 = sW[0][c], w1 = sW[1][c], w2 = sW[2][c], w3 = sW[3][c];
        #pragma unroll
        for (int v = 0; v < 8; ++v) {
            float xg = sPn[c][v + 8 * half] - pmb;
            float g = 0.5f * xg * (1.0f + erff(xg * 0.70710678118654752f));
            pos0[v] = fmaf(w0, g, pos0[v]);
            pos1[v] = fmaf(w1, g, pos1[v]);
            pos2[v] = fmaf(w2, g, pos2[v]);
            pos3[v] = fmaf(w3, g, pos3[v]);
        }
    }

    const float scale = 0.17677669529663688f;            // 1/sqrt(32)
    #pragma unroll
    for (int h = 0; h < HEADS; ++h) {
        float bb = bbar[h];
        const float* ph = (h == 0) ? pos0 : (h == 1) ? pos1 : (h == 2) ? pos2 : pos3;
        #pragma unroll
        for (int v = 0; v < 8; ++v) {
            int n = n0 + v + 8 * half;
            int m = m0 + lo;
            scores[((size_t)h * NPTS + n) * NPTS + m] = acc[h][v] * scale + ph[v] + bb;
        }
    }
}

// ---------------- row softmax -> bf16 attn ----------------
__global__ __launch_bounds__(128) void k_softmax(const float* scores, __bf16* attn) {
    int row = blockIdx.x;                                // h*N + n
    const float* s = scores + (size_t)row * NPTS;
    __bf16* a = attn + (size_t)row * NPTS;
    int tid = threadIdx.x;
    __shared__ float redm[4], reds[4];
    float v[8];
    float mx = -INFINITY;
    #pragma unroll
    for (int i = 0; i < 8; ++i) { v[i] = s[tid + i * 128]; mx = fmaxf(mx, v[i]); }
    for (int off = 16; off > 0; off >>= 1) mx = fmaxf(mx, __shfl_xor(mx, off, 32));
    if ((tid & 31) == 0) redm[tid >> 5] = mx;
    __syncthreads();
    mx = fmaxf(fmaxf(redm[0], redm[1]), fmaxf(redm[2], redm[3]));
    float sum = 0.f;
    #pragma unroll
    for (int i = 0; i < 8; ++i) { v[i] = __expf(v[i] - mx); sum += v[i]; }
    for (int off = 16; off > 0; off >>= 1) sum += __shfl_xor(sum, off, 32);
    if ((tid & 31) == 0) reds[tid >> 5] = sum;
    __syncthreads();
    float inv = 1.0f / (reds[0] + reds[1] + reds[2] + reds[3]);
    #pragma unroll
    for (int i = 0; i < 8; ++i) a[tid + i * 128] = (__bf16)(v[i] * inv);
}

// ---------------- attn . V (WMMA) -> obuf[n][c] bf16 ----------------
__global__ __launch_bounds__(32) void k_av(const __bf16* attn, const __bf16* vb,
                                           __bf16* obuf) {
    int nt = blockIdx.x;                                 // 0..63
    int ct = blockIdx.y;                                 // 0..7  -> c tile (h = ct>>1)
    int lane = threadIdx.x, half = lane >> 4, lo = lane & 15;
    int h = ct >> 1;
    const __bf16* arow = attn + ((size_t)h * NPTS + nt * 16 + lo) * NPTS;  // attn[h][n][*]
    const __bf16* brow = vb + (size_t)(ct * 16 + lo) * NPTS;               // V[c][*]
    v8f acc = {};
    for (int k0 = 0; k0 < NPTS; k0 += 32) {
        if (k0 + 256 < NPTS) {                           // global_prefetch_b8 ahead of stream
            __builtin_prefetch((const void*)(arow + k0 + 256), 0, 1);
            __builtin_prefetch((const void*)(brow + k0 + 256), 0, 1);
        }
        v16bf a = load_afrag(arow + k0, half);
        v16bf b = load_bfrag(brow + k0, half);
        acc = wmma_bf16(a, b, acc);
    }
    #pragma unroll
    for (int v = 0; v < 8; ++v) {
        int n = nt * 16 + v + 8 * half;
        int c = ct * 16 + lo;
        obuf[n * CCH + c] = (__bf16)acc[v];              // [n][c] layout = B operand of Wo GEMM
    }
}

// ---------------- Wo GEMM + bias + residual -> y fp32 [C][N] ----------------
__global__ __launch_bounds__(32) void k_ogemm(const __bf16* wo, const __bf16* obuf,
                                              const float* bo, const float* x, float* y) {
    int ot = blockIdx.x, nt = blockIdx.y;
    int lane = threadIdx.x, half = lane >> 4, lo = lane & 15;
    int arow = ot * 16 + lo;
    int brow = nt * 16 + lo;
    v8f acc = {};
    #pragma unroll
    for (int k0 = 0; k0 < CCH; k0 += 32) {
        v16bf a = load_afrag(wo + arow * CCH + k0, half);
        v16bf b = load_bfrag(obuf + brow * CCH + k0, half);
        acc = wmma_bf16(a, b, acc);
    }
    #pragma unroll
    for (int v = 0; v < 8; ++v) {
        int o = ot * 16 + v + 8 * half;
        int n = nt * 16 + lo;
        y[o * NPTS + n] = acc[v] + bo[o] + x[o * NPTS + n];
    }
}

// ---------------- channel LayerNorm per point ----------------
__global__ __launch_bounds__(32) void k_ln(const float* y, const float* gamma,
                                           const float* beta, float* out) {
    int n = blockIdx.x;
    int lane = threadIdx.x;
    float vals[4];
    float s = 0.f, s2 = 0.f;
    #pragma unroll
    for (int k = 0; k < 4; ++k) {
        float t = y[(lane + 32 * k) * NPTS + n];
        vals[k] = t; s += t; s2 += t * t;
    }
    for (int off = 16; off > 0; off >>= 1) {
        s  += __shfl_xor(s,  off, 32);
        s2 += __shfl_xor(s2, off, 32);
    }
    float mu  = s * (1.f / 128.f);
    float var = s2 * (1.f / 128.f) - mu * mu;
    float r = rsqrtf(var + 1e-5f);
    #pragma unroll
    for (int k = 0; k < 4; ++k) {
        int c = lane + 32 * k;
        out[c * NPTS + n] = (vals[k] - mu) * r * gamma[c] + beta[c];
    }
}

// ---------------- launcher ----------------
extern "C" void kernel_launch(void* const* d_in, const int* in_sizes, int n_in,
                              void* d_out, int out_size, void* d_ws, size_t ws_size,
                              hipStream_t stream) {
    const float* x    = (const float*)d_in[0];
    const float* xyz  = (const float*)d_in[1];
    const float* Wq   = (const float*)d_in[2];
    const float* Wk   = (const float*)d_in[3];
    const float* Wv   = (const float*)d_in[4];
    const float* Wp1  = (const float*)d_in[5];
    const float* bp1  = (const float*)d_in[6];
    const float* Wp2  = (const float*)d_in[7];
    const float* bp2  = (const float*)d_in[8];
    const float* Wo   = (const float*)d_in[9];
    const float* bo   = (const float*)d_in[10];
    const float* gam  = (const float*)d_in[11];
    const float* bet  = (const float*)d_in[12];
    float* out = (float*)d_out;

    char* w = (char*)d_ws;
    size_t off = 0;
    auto take = [&](size_t bytes) { char* p = w + off; off = (off + bytes + 255) & ~(size_t)255; return p; };

    __bf16* wbf    = (__bf16*)take(4 * CCH * CCH * sizeof(__bf16));   // wq|wk|wv|wo
    __bf16* xt     = (__bf16*)take((size_t)NPTS * CCH * sizeof(__bf16));
    float*  P      = (float*) take((size_t)CCH * NPTS * sizeof(float));
    float*  wbar   = (float*) take(HEADS * CCH * sizeof(float));
    float*  bbar   = (float*) take(HEADS * sizeof(float));
    __bf16* qh     = (__bf16*)take((size_t)CCH * NPTS * sizeof(__bf16));
    __bf16* kh     = (__bf16*)take((size_t)CCH * NPTS * sizeof(__bf16));
    __bf16* vb     = (__bf16*)take((size_t)CCH * NPTS * sizeof(__bf16));
    float*  scores = (float*) take((size_t)HEADS * NPTS * NPTS * sizeof(float));
    __bf16* attn   = (__bf16*)take((size_t)HEADS * NPTS * NPTS * sizeof(__bf16));
    __bf16* obuf   = (__bf16*)take((size_t)NPTS * CCH * sizeof(__bf16));
    float*  ybuf   = (float*) take((size_t)CCH * NPTS * sizeof(float));
    (void)ws_size; (void)n_in; (void)in_sizes; (void)out_size;

    __bf16* wq_bf = wbf;
    __bf16* wk_bf = wbf + 16384;
    __bf16* wv_bf = wbf + 32768;
    __bf16* wo_bf = wbf + 49152;

    k_cvt_w <<<256, 256, 0, stream>>>(Wq, Wk, Wv, Wo, wbf);
    k_xtP   <<<512, 256, 0, stream>>>(x, xyz, Wp1, xt, P);
    k_wbar  <<<2,   256, 0, stream>>>(Wp2, bp2, wbar, bbar);

    k_gemm_qkv<<<dim3(8, 64, 3), 32, 0, stream>>>(wq_bf, wk_bf, wv_bf, xt, qh, kh, vb);
    k_scores  <<<dim3(64, 64),    32, 0, stream>>>(qh, kh, P, bp1, wbar, bbar, scores);
    k_softmax <<<HEADS * NPTS,   128, 0, stream>>>(scores, attn);
    k_av      <<<dim3(64, 8),     32, 0, stream>>>(attn, vb, obuf);
    k_ogemm   <<<dim3(8, 64),     32, 0, stream>>>(wo_bf, obuf, bo, x, ybuf);
    k_ln      <<<NPTS,            32, 0, stream>>>(ybuf, gam, bet, out);
}
